// MAGPoolGCN_39865886442008
// MI455X (gfx1250) — matrix operand order repaired and checked by reading, not compiled, not run on hardware
//
#include <hip/hip_runtime.h>
#include <hip/hip_bf16.h>
#include <math.h>

#define NB 32          // graphs
#define HID 128
#define NE 524288      // edges

typedef __attribute__((ext_vector_type(16))) _Float16 v16h;
typedef __attribute__((ext_vector_type(8)))  float    v8f;

// ---------------- utility kernels ----------------
__global__ void k_fill_f32(float* __restrict__ p, float v, int n) {
  int i = blockIdx.x * blockDim.x + threadIdx.x;
  if (i < n) p[i] = v;
}
__global__ void k_copy_i32(int* __restrict__ d, const int* __restrict__ s, int n) {
  int i = blockIdx.x * blockDim.x + threadIdx.x;
  if (i < n) d[i] = s[i];
}

// ---------------- block-diagonal projection via WMMA ----------------
// h[n, h*32+e] = sum_d x[n, h*32+d] * W[h][d][e]   (W: [4][32][32] f32)
// One wave computes one 16x16 output tile with a single v_wmma_f32_16x16x32_f16.
__global__ void k_proj_wmma(const float* __restrict__ x, const float* __restrict__ W,
                            float* __restrict__ out, int n) {
  int wave = (blockIdx.x * blockDim.x + threadIdx.x) >> 5;
  int lane = threadIdx.x & 31;
  int ntiles = (n >> 4) * 8;            // (n/16) row blocks * 4 heads * 2 col tiles
  if (wave >= ntiles) return;           // wave-uniform exit (EXEC stays all-ones)
  int rowblk = wave >> 3;
  int rem    = wave & 7;
  int head   = rem >> 1;
  int ct     = rem & 1;

  // ---- A fragment: 16x32 f16, CDNA5 layout ----
  int row   = (rowblk << 4) + (lane & 15);
  int kbase = (lane < 16) ? 0 : 8;
  const float* xr = x + (size_t)row * HID + head * 32;
  v16h a;
  #pragma unroll
  for (int v = 0; v < 8; ++v) {
    int kp = ((v < 4) ? (2 * v) : (16 + 2 * (v - 4))) + kbase;  // even -> 8B aligned
    float2 xx = *(const float2*)(xr + kp);
    a[2 * v]     = (_Float16)xx.x;
    a[2 * v + 1] = (_Float16)xx.y;
  }

  // ---- B fragment: 32x16 f16, lane = column, same K order as A ----
  const float* Wh = W + head * 32 * 32;
  int col = (ct << 4) + (lane & 15);
  int kb2 = (lane >> 4) << 3;
  v16h b;
  #pragma unroll
  for (int v = 0; v < 8; ++v) {
    int k0 = ((v < 4) ? (2 * v) : (16 + 2 * (v - 4))) + kb2;
    b[2 * v]     = (_Float16)Wh[k0 * 32 + col];
    b[2 * v + 1] = (_Float16)Wh[(k0 + 1) * 32 + col];
  }

  v8f c = {};
  c = __builtin_amdgcn_wmma_f32_16x16x32_f16(false, a, false, b, (short)0, c, false, false);

  // ---- D: VGPR r -> M = r + 8*(lane/16), N = lane&15 ----
  int mbase = (lane >> 4) << 3;
  int nn = lane & 15;
  #pragma unroll
  for (int r = 0; r < 8; ++r) {
    int m = mbase + r;
    out[(size_t)((rowblk << 4) + m) * HID + head * 32 + (ct << 4) + nn] = c[r];
  }
}

// ---------------- degree / rsqrt ----------------
__global__ void k_deg_edges(const int* __restrict__ dst, const float* __restrict__ w,
                            float* __restrict__ deg, int e) {
  int i = blockIdx.x * blockDim.x + threadIdx.x;
  if (i >= e) return;
  float we = w[i];
  if (we != 0.0f) atomicAdd(&deg[dst[i]], we);
}
__global__ void k_rsqrt(float* __restrict__ deg, int n) {
  int i = blockIdx.x * blockDim.x + threadIdx.x;
  if (i < n) deg[i] = rsqrtf(deg[i]);   // deg -> dinv in place
}

// ---------------- edge feature aggregation (segment_sum) ----------------
__global__ void k_agg_feat(const int* __restrict__ src, const int* __restrict__ dst,
                           const float* __restrict__ w, const float* __restrict__ dinv,
                           const float* __restrict__ h, float* __restrict__ agg, int e) {
  int gid = blockIdx.x * blockDim.x + threadIdx.x;
  int ei = gid >> 5;
  if (ei >= e) return;
  float we = w[ei];
  if (we == 0.0f) return;
  int c = (gid & 31) << 2;
  int s = src[ei], d = dst[ei];
  float coef = dinv[s] * dinv[d] * we;
  float4 hv = *(const float4*)(h + (size_t)s * HID + c);
  float* ad = agg + (size_t)d * HID + c;
  atomicAdd(ad + 0, hv.x * coef);
  atomicAdd(ad + 1, hv.y * coef);
  atomicAdd(ad + 2, hv.z * coef);
  atomicAdd(ad + 3, hv.w * coef);
}

// ---------------- mh_gcn epilogue: self-loop + bias + relu ----------------
__global__ void k_combine(const float* __restrict__ agg, const float* __restrict__ hp,
                          const float* __restrict__ dinv, const float* __restrict__ bias,
                          float* __restrict__ out, int n) {
  int gid = blockIdx.x * blockDim.x + threadIdx.x;
  if (gid >= n * HID) return;
  int i = gid >> 7, f = gid & 127;
  float di = dinv[i];
  float v = agg[gid] + hp[gid] * di * di + bias[f];
  out[gid] = fmaxf(v, 0.0f);
}

// ---------------- attention: t = (att * sub) . psW per node ----------------
__global__ void k_att_t(const float* __restrict__ h, const float* __restrict__ A,
                        const float* __restrict__ psW, float* __restrict__ t, int n) {
  int i = blockIdx.x * blockDim.x + threadIdx.x;
  if (i >= n) return;
  const float* hr = h + (size_t)i * HID;
  float acc = 0.0f;
  #pragma unroll
  for (int hd = 0; hd < 4; ++hd) {
    float att = 0.0f;
    for (int d = 0; d < 32; ++d) att += hr[hd * 32 + d] * A[hd * 32 + d];
    float s = 0.0f;
    for (int d = 0; d < 32; ++d) s += hr[hd * 32 + d] * psW[hd * 32 + d];
    acc += att * s;
  }
  t[i] = acc;
}

__global__ void k_agg_scal(const int* __restrict__ src, const int* __restrict__ dst,
                           const float* __restrict__ w, const float* __restrict__ dinv,
                           const float* __restrict__ t, float* __restrict__ tagg, int e) {
  int i = blockIdx.x * blockDim.x + threadIdx.x;
  if (i >= e) return;
  float we = w[i];
  if (we == 0.0f) return;
  int s = src[i], d = dst[i];
  atomicAdd(&tagg[d], t[s] * dinv[s] * dinv[d] * we);
}

__global__ void k_score(const float* __restrict__ tagg, const float* __restrict__ t,
                        const float* __restrict__ dinv, const float* __restrict__ psb,
                        float* __restrict__ score, int n) {
  int i = blockIdx.x * blockDim.x + threadIdx.x;
  if (i >= n) return;
  float di = dinv[i];
  score[i] = tagg[i] + t[i] * di * di + psb[0];
}

// ---------------- per-graph top-k by rank counting ----------------
__global__ void k_topk(const float* __restrict__ score, int* __restrict__ nmap,
                       int* __restrict__ perm, float* __restrict__ wsel,
                       int nper, int k) {
  __shared__ float s[1024];
  int g = blockIdx.x;
  const float* sg = score + (size_t)g * nper;
  for (int i = threadIdx.x; i < nper; i += blockDim.x) s[i] = sg[i];
  __syncthreads();
  for (int i = threadIdx.x; i < nper; i += blockDim.x) {
    float si = s[i];
    int r = 0;
    for (int j = 0; j < nper; ++j) {
      float sj = s[j];
      r += (sj > si) || (sj == si && j < i);
    }
    int old = g * nper + i;
    if (r < k) {
      int ni = g * k + r;
      perm[ni] = old;
      wsel[ni] = tanhf(si);
      nmap[old] = ni;
    } else {
      nmap[old] = -1;
    }
  }
}

__global__ void k_gather(const float* __restrict__ h, const int* __restrict__ perm,
                         const float* __restrict__ wsel, float* __restrict__ out, int nk) {
  int gid = blockIdx.x * blockDim.x + threadIdx.x;
  int node = gid >> 5;
  if (node >= nk) return;
  int c = (gid & 31) << 2;
  int old = perm[node];
  float ww = wsel[node];
  float4 v = *(const float4*)(h + (size_t)old * HID + c);
  float4 o; o.x = v.x * ww; o.y = v.y * ww; o.z = v.z * ww; o.w = v.w * ww;
  *(float4*)(out + (size_t)node * HID + c) = o;
}

// ---------------- readout: max + mean, accumulated into r ----------------
__global__ void k_readout(const float* __restrict__ xp, float* __restrict__ r, int k) {
  int tid = blockIdx.x * blockDim.x + threadIdx.x;
  if (tid >= NB * HID) return;
  int g = tid >> 7, f = tid & 127;
  float mx = -3.402823466e+38f, sm = 0.0f;
  const float* base = xp + (size_t)g * k * HID + f;
  for (int i = 0; i < k; ++i) {
    float v = base[(size_t)i * HID];
    mx = fmaxf(mx, v);
    sm += v;
  }
  r[g * 256 + f]       += mx;
  r[g * 256 + 128 + f] += sm * (1.0f / k);
}

// ---------------- edge filter / remap ----------------
__global__ void k_filter(const int* __restrict__ src, const int* __restrict__ dst,
                         const float* __restrict__ w, const int* __restrict__ nmap,
                         int* __restrict__ sn, int* __restrict__ dn,
                         float* __restrict__ wn, int e) {
  int i = blockIdx.x * blockDim.x + threadIdx.x;
  if (i >= e) return;
  int ns = nmap[src[i]], nd = nmap[dst[i]];
  float we = (ns >= 0 && nd >= 0) ? w[i] : 0.0f;
  sn[i] = ns > 0 ? ns : 0;
  dn[i] = nd > 0 ? nd : 0;
  wn[i] = we;
}

// ---------------- final MLP + log_softmax (single workgroup) ----------------
__global__ void k_mlp(const float* __restrict__ r,
                      const float* __restrict__ l1W, const float* __restrict__ l1b,
                      const float* __restrict__ l2W, const float* __restrict__ l2b,
                      const float* __restrict__ l3W, const float* __restrict__ l3b,
                      float* __restrict__ out) {
  __shared__ float z1[32 * 128];
  __shared__ float z2[32 * 64];
  __shared__ float z3[32 * 10];
  int tid = threadIdx.x;
  for (int idx = tid; idx < 32 * 128; idx += blockDim.x) {
    int row = idx >> 7, col = idx & 127;
    float acc = l1b[col];
    for (int k = 0; k < 256; ++k) acc += r[row * 256 + k] * l1W[k * 128 + col];
    z1[idx] = fmaxf(acc, 0.0f);
  }
  __syncthreads();
  for (int idx = tid; idx < 32 * 64; idx += blockDim.x) {
    int row = idx >> 6, col = idx & 63;
    float acc = l2b[col];
    for (int k = 0; k < 128; ++k) acc += z1[row * 128 + k] * l2W[k * 64 + col];
    z2[idx] = fmaxf(acc, 0.0f);
  }
  __syncthreads();
  for (int idx = tid; idx < 32 * 10; idx += blockDim.x) {
    int row = idx / 10, col = idx % 10;
    float acc = l3b[col];
    for (int k = 0; k < 64; ++k) acc += z2[row * 64 + k] * l3W[k * 10 + col];
    z3[idx] = acc;
  }
  __syncthreads();
  if (tid < 32) {
    float m = -3.402823466e+38f;
    for (int c = 0; c < 10; ++c) m = fmaxf(m, z3[tid * 10 + c]);
    float s = 0.0f;
    for (int c = 0; c < 10; ++c) s += expf(z3[tid * 10 + c] - m);
    float lse = m + logf(s);
    for (int c = 0; c < 10; ++c) out[tid * 10 + c] = z3[tid * 10 + c] - lse;
  }
}

// =====================================================================
extern "C" void kernel_launch(void* const* d_in, const int* in_sizes, int n_in,
                              void* d_out, int out_size, void* d_ws, size_t ws_size,
                              hipStream_t stream) {
  (void)in_sizes; (void)n_in; (void)out_size; (void)ws_size;
  const int N0 = NB * 1024;

  const float* x   = (const float*)d_in[0];
  const int*   src = (const int*)d_in[1];
  const int*   dst = (const int*)d_in[2];
  const float* Ws[3]   = {(const float*)d_in[3],  (const float*)d_in[8],  (const float*)d_in[13]};
  const float* bs[3]   = {(const float*)d_in[4],  (const float*)d_in[9],  (const float*)d_in[14]};
  const float* As[3]   = {(const float*)d_in[5],  (const float*)d_in[10], (const float*)d_in[15]};
  const float* psWs[3] = {(const float*)d_in[6],  (const float*)d_in[11], (const float*)d_in[16]};
  const float* psbs[3] = {(const float*)d_in[7],  (const float*)d_in[12], (const float*)d_in[17]};
  const float* l1W = (const float*)d_in[18]; const float* l1b = (const float*)d_in[19];
  const float* l2W = (const float*)d_in[20]; const float* l2b = (const float*)d_in[21];
  const float* l3W = (const float*)d_in[22]; const float* l3b = (const float*)d_in[23];

  // ---- workspace bump allocator (256B aligned) ----
  char* base = (char*)d_ws; size_t off = 0;
  auto alloc = [&](size_t bytes) -> void* {
    void* p = base + off; off = (off + bytes + 255) & ~(size_t)255; return p;
  };
  float* bufP = (float*)alloc((size_t)N0 * HID * 4);        // projection
  float* bufG = (float*)alloc((size_t)N0 * HID * 4);        // edge aggregate
  float* bufH = (float*)alloc((size_t)N0 * HID * 4);        // layer output
  float* bufX = (float*)alloc((size_t)NB * 512 * HID * 4);  // pooled features
  float* deg  = (float*)alloc((size_t)N0 * 4);              // deg -> dinv
  float* tbuf = (float*)alloc((size_t)N0 * 4);
  float* tagg = (float*)alloc((size_t)N0 * 4);
  float* scor = (float*)alloc((size_t)N0 * 4);
  int*   nmap = (int*)  alloc((size_t)N0 * 4);
  int*   perm = (int*)  alloc((size_t)NB * 512 * 4);
  float* wsel = (float*)alloc((size_t)NB * 512 * 4);
  int*   sA = (int*)alloc((size_t)NE * 4);
  int*   dA = (int*)alloc((size_t)NE * 4);
  float* wA = (float*)alloc((size_t)NE * 4);
  int*   sB = (int*)alloc((size_t)NE * 4);
  int*   dB = (int*)alloc((size_t)NE * 4);
  float* wB = (float*)alloc((size_t)NE * 4);
  float* rbuf = (float*)alloc((size_t)NB * 256 * 4);

  const int TB = 256;
  #define GRID(n) dim3((unsigned)(((size_t)(n) + TB - 1) / TB))

  // ---- init: level-0 edge list, w = 1, r = 0 ----
  k_copy_i32<<<GRID(NE), TB, 0, stream>>>(sA, src, NE);
  k_copy_i32<<<GRID(NE), TB, 0, stream>>>(dA, dst, NE);
  k_fill_f32<<<GRID(NE), TB, 0, stream>>>(wA, 1.0f, NE);
  k_fill_f32<<<GRID(NB * 256), TB, 0, stream>>>(rbuf, 0.0f, NB * 256);

  int ns[3]  = {N0, NB * 512, NB * 256};
  int ksz[3] = {512, 256, 128};

  const float* hin = x;
  int *scur = sA, *dcur = dA; float* wcur = wA;
  int *snxt = sB, *dnxt = dB; float* wnxt = wB;

  for (int l = 0; l < 3; ++l) {
    int n = ns[l], k = ksz[l], nper = n / NB;

    // degrees -> dinv
    k_fill_f32<<<GRID(n), TB, 0, stream>>>(deg, 1.0f, n);
    k_deg_edges<<<GRID(NE), TB, 0, stream>>>(dcur, wcur, deg, NE);
    k_rsqrt<<<GRID(n), TB, 0, stream>>>(deg, n);

    // block-diagonal projection (WMMA)
    int tiles = (n >> 4) * 8;
    k_proj_wmma<<<GRID((size_t)tiles * 32), TB, 0, stream>>>(hin, Ws[l], bufP, n);

    // symmetric-normalized edge aggregation
    k_fill_f32<<<GRID((size_t)n * HID), TB, 0, stream>>>(bufG, 0.0f, n * HID);
    k_agg_feat<<<GRID((size_t)NE * 32), TB, 0, stream>>>(scur, dcur, wcur, deg, bufP, bufG, NE);
    k_combine<<<GRID((size_t)n * HID), TB, 0, stream>>>(bufG, bufP, deg, bs[l], bufH, n);

    // attention scores
    k_att_t<<<GRID(n), TB, 0, stream>>>(bufH, As[l], psWs[l], tbuf, n);
    k_fill_f32<<<GRID(n), TB, 0, stream>>>(tagg, 0.0f, n);
    k_agg_scal<<<GRID(NE), TB, 0, stream>>>(scur, dcur, wcur, deg, tbuf, tagg, NE);
    k_score<<<GRID(n), TB, 0, stream>>>(tagg, tbuf, deg, psbs[l], scor, n);

    // top-k pooling, gather, readout
    k_topk<<<NB, TB, 0, stream>>>(scor, nmap, perm, wsel, nper, k);
    int nk = NB * k;
    k_gather<<<GRID((size_t)nk * 32), TB, 0, stream>>>(bufH, perm, wsel, bufX, nk);
    k_readout<<<GRID(NB * HID), TB, 0, stream>>>(bufX, rbuf, k);

    // edge remap (not needed after last level)
    if (l < 2) {
      k_filter<<<GRID(NE), TB, 0, stream>>>(scur, dcur, wcur, nmap, snxt, dnxt, wnxt, NE);
      int* ti = scur; scur = snxt; snxt = ti;
      ti = dcur; dcur = dnxt; dnxt = ti;
      float* tf = wcur; wcur = wnxt; wnxt = tf;
    }
    hin = bufX;
  }

  // classifier head
  k_mlp<<<1, TB, 0, stream>>>(rbuf, l1W, l1b, l2W, l2b, l3W, l3b, (float*)d_out);
  #undef GRID
}
